// EGNNLayer_62783831933162
// MI455X (gfx1250) — matrix operand order repaired and checked by reading, not compiled, not run on hardware
//
#include <hip/hip_runtime.h>
#include <hip/hip_bf16.h>

typedef __attribute__((ext_vector_type(16))) _Float16 v16h;
typedef __attribute__((ext_vector_type(8)))  _Float16 v8h;
typedef __attribute__((ext_vector_type(4)))  _Float16 v4h;
typedef __attribute__((ext_vector_type(8)))  float    v8f;

#define F 128
#define H 128
#define WAVES 4   // waves per block, 16 rows per wave tile

__device__ __forceinline__ float silu_f(float x) {
    return x / (1.0f + __expf(-x));
}

__device__ __forceinline__ v16h make_v16(v8h lo, v8h hi) {
    union { v16h v; v8h h[2]; } u;
    u.h[0] = lo; u.h[1] = hi;
    return u.v;
}

// ---------------------------------------------------------------------------
// Wave-level GEMM: D[16 x 128] += A[16 x 32*KSTEPS] * B, f16 in / f32 acc.
// A rows live in LDS (rowStride halfs per row), laid out per the CDNA5 16-bit
// A-matrix table: lane<16 holds K 0-7 & 16-23 of its 32-slice, lane>=16 holds
// K 8-15 & 24-31.  B is pre-packed: per (kstep,ntile) fragment, lane = N column
// (+16 for K 8-15 & 24-31), 16 K halfs per lane in A's order.
// ---------------------------------------------------------------------------
template<int KSTEPS>
__device__ __forceinline__ void wave_gemm(const _Float16* aBuf, int rowStride,
                                          const _Float16* bPack,
                                          v8f acc[8], int lane) {
    const int m     = lane & 15;
    const int hiSel = (lane >> 4) ? 8 : 0;
#pragma unroll
    for (int ks = 0; ks < KSTEPS; ++ks) {
        const _Float16* ap = aBuf + m * rowStride + ks * 32 + hiSel;
        v8h lo = *(const v8h*)(ap);
        v8h hi = *(const v8h*)(ap + 16);
        v16h a = make_v16(lo, hi);
#pragma unroll
        for (int nt = 0; nt < 8; ++nt) {
            v16h b = *(const v16h*)(bPack + ((size_t)(ks * 8 + nt) * 32 + lane) * 16);
            acc[nt] = __builtin_amdgcn_wmma_f32_16x16x32_f16(
                false, a, false, b, (short)0, acc[nt], false, false);
        }
    }
}

// ---------------------------------------------------------------------------
// Init: zero the fp32 aggregation buffer, seed pos_out with pos.
// ---------------------------------------------------------------------------
__global__ void egnn_init_kernel(float* __restrict__ agg,
                                 float* __restrict__ posOut,
                                 const float* __restrict__ pos,
                                 int N) {
    int idx = blockIdx.x * blockDim.x + threadIdx.x;
    if (idx < N * F) agg[idx] = 0.0f;
    if (idx < N * 3) posOut[idx] = pos[idx];
}

// ---------------------------------------------------------------------------
// Pack a fp32 weight matrix W[K][128] into WMMA-B f16 fragments.
// ---------------------------------------------------------------------------
__global__ void egnn_pack_w16(const float* __restrict__ src,
                              _Float16* __restrict__ dst, int K) {
    int idx = blockIdx.x * blockDim.x + threadIdx.x;
    if (idx >= K * 128) return;
    int k = idx >> 7;
    int n = idx & 127;
    int frag = (k >> 5) * 8 + (n >> 4);
    int off  = (frag * 32 + (n & 15) + 16 * ((k >> 3) & 1)) * 16 + (k & 7) + (((k >> 4) & 1) << 3);
    dst[off] = (_Float16)src[idx];
}

// ---------------------------------------------------------------------------
// Fused edge pipeline: radial, edge MLP (2 GEMMs + SiLU), scatter-add of msg
// into agg[receivers], pos-correction MLP (GEMM + SiLU + dot with pW2),
// scatter-add of clipped trans into pos_out[senders].
// ---------------------------------------------------------------------------
__global__ __launch_bounds__(WAVES * 32)
void egnn_edge_kernel(const float* __restrict__ h,
                      const float* __restrict__ pos,
                      const int*   __restrict__ senders,
                      const int*   __restrict__ receivers,
                      const float* __restrict__ eW1,   // [257][128] fp32 (row 256 = radial)
                      const float* __restrict__ eb1,
                      const float* __restrict__ eb2,
                      const float* __restrict__ pb1,
                      const float* __restrict__ pW2,   // [128] fp32
                      const _Float16* __restrict__ packE1,
                      const _Float16* __restrict__ packE2,
                      const _Float16* __restrict__ packP1,
                      float* __restrict__ agg,
                      float* __restrict__ posOut,
                      int E, int nTiles) {
    __shared__ __align__(16) _Float16 sIn[WAVES][16][2 * F];  // gathered feats; reused as msg
    __shared__ __align__(16) _Float16 sHid[WAVES][16][H];
    __shared__ float sRad[WAVES][16];
    __shared__ float sCd[WAVES][16][3];
    __shared__ float sPc[WAVES][16];
    __shared__ int   sSend[WAVES][16];
    __shared__ int   sRecv[WAVES][16];

    const int lane = threadIdx.x & 31;
    const int w    = threadIdx.x >> 5;
    const int tile = blockIdx.x * WAVES + w;
    const bool valid = tile < nTiles;
    const int e0 = tile * 16;
    const int hiM = (lane >> 4) * 8;
    const int nl  = lane & 15;

    // ---- stage 1: edge meta (indices, coord_diff, radial) ----
    if (valid && lane < 16) {
        int e = e0 + lane;
        if (e < E) {
            int s = senders[e];
            int r = receivers[e];
            sSend[w][lane] = s;
            sRecv[w][lane] = r;
            float dx = pos[(size_t)s * 3 + 0] - pos[(size_t)r * 3 + 0];
            float dy = pos[(size_t)s * 3 + 1] - pos[(size_t)r * 3 + 1];
            float dz = pos[(size_t)s * 3 + 2] - pos[(size_t)r * 3 + 2];
            sCd[w][lane][0] = dx; sCd[w][lane][1] = dy; sCd[w][lane][2] = dz;
            sRad[w][lane] = dx * dx + dy * dy + dz * dz;
        } else {
            sSend[w][lane] = -1;
            sRecv[w][lane] = -1;
            sCd[w][lane][0] = sCd[w][lane][1] = sCd[w][lane][2] = 0.0f;
            sRad[w][lane] = 0.0f;
        }
        sPc[w][lane] = 0.0f;
    }
    if (valid && lane == 0) {
        __builtin_prefetch(packE1, 0, 3);   // global_prefetch_b8: pull weights toward L2
        __builtin_prefetch(packE2, 0, 3);
        __builtin_prefetch(packP1, 0, 3);
    }
    __syncthreads();

    // ---- stage 2: gather h[senders] | h[receivers] -> f16 LDS [16][256] ----
    if (valid) {
#pragma unroll 4
        for (int m = 0; m < 16; ++m) {
            int s = sSend[w][m];
            int r = sRecv[w][m];
            v4h za, zb;
            if (s >= 0) {
                float4 a = ((const float4*)(h + (size_t)s * F))[lane];
                float4 b = ((const float4*)(h + (size_t)r * F))[lane];
                za = (v4h){(_Float16)a.x, (_Float16)a.y, (_Float16)a.z, (_Float16)a.w};
                zb = (v4h){(_Float16)b.x, (_Float16)b.y, (_Float16)b.z, (_Float16)b.w};
            } else {
                za = (v4h)(_Float16)0.0f;
                zb = za;
            }
            *(v4h*)&sIn[w][m][lane * 4]     = za;
            *(v4h*)&sIn[w][m][F + lane * 4] = zb;
        }
    }
    __syncthreads();

    // ---- stage 3: edge GEMM1 (K=256 via WMMA) + radial column + bias + SiLU ----
    if (valid) {
        v8f acc[8];
#pragma unroll
        for (int nt = 0; nt < 8; ++nt) acc[nt] = (v8f)0.0f;
        wave_gemm<8>(&sIn[w][0][0], 2 * F, packE1, acc, lane);
#pragma unroll
        for (int nt = 0; nt < 8; ++nt) {
            int n = nt * 16 + nl;
            float w256 = eW1[(size_t)(2 * F) * H + n];   // row 256: radial weight
            float b1   = eb1[n];
#pragma unroll
            for (int j = 0; j < 8; ++j) {
                int m = j + hiM;
                float x = acc[nt][j] + sRad[w][m] * w256 + b1;
                sHid[w][m][n] = (_Float16)silu_f(x);
            }
        }
    }
    __syncthreads();

    // ---- stage 4: edge GEMM2 + SiLU -> msg; scatter-add into agg[recv] ----
    if (valid) {
        v8f acc[8];
#pragma unroll
        for (int nt = 0; nt < 8; ++nt) acc[nt] = (v8f)0.0f;
        wave_gemm<4>(&sHid[w][0][0], H, packE2, acc, lane);
#pragma unroll
        for (int nt = 0; nt < 8; ++nt) {
            int n = nt * 16 + nl;
            float b2 = eb2[n];
#pragma unroll
            for (int j = 0; j < 8; ++j) {
                int m = j + hiM;
                float msg = silu_f(acc[nt][j] + b2);
                int r = sRecv[w][m];
                if (r >= 0)
                    unsafeAtomicAdd(&agg[(size_t)r * H + n], msg);
                sIn[w][m][n] = (_Float16)msg;   // reuse sIn (stride 2F) as msg buffer
            }
        }
    }
    __syncthreads();

    // ---- stage 5: pos MLP GEMM (pW1) + SiLU, dot with pW2 -> pc[m] ----
    if (valid) {
        v8f acc[8];
#pragma unroll
        for (int nt = 0; nt < 8; ++nt) acc[nt] = (v8f)0.0f;
        wave_gemm<4>(&sIn[w][0][0], 2 * F, packP1, acc, lane);
        float pacc[8];
#pragma unroll
        for (int j = 0; j < 8; ++j) pacc[j] = 0.0f;
#pragma unroll
        for (int nt = 0; nt < 8; ++nt) {
            int n = nt * 16 + nl;
            float pb  = pb1[n];
            float w2  = pW2[n];
#pragma unroll
            for (int j = 0; j < 8; ++j)
                pacc[j] += silu_f(acc[nt][j] + pb) * w2;
        }
#pragma unroll
        for (int j = 0; j < 8; ++j)
            atomicAdd(&sPc[w][j + hiM], pacc[j]);   // ds_add_f32
    }
    __syncthreads();

    // ---- stage 6: trans = clip(coord_diff * pc); scatter-add into pos_out ----
    if (valid && lane < 16) {
        int s = sSend[w][lane];
        if (s >= 0) {
            float pc = sPc[w][lane];
#pragma unroll
            for (int i = 0; i < 3; ++i) {
                float t = sCd[w][lane][i] * pc;
                t = fminf(fmaxf(t, -100.0f), 100.0f);
                unsafeAtomicAdd(&posOut[(size_t)s * 3 + i], t);
            }
        }
    }
}

// ---------------------------------------------------------------------------
// Node pipeline: concat(h, agg) -> node MLP (2 GEMMs, SiLU on first) + residual.
// ---------------------------------------------------------------------------
__global__ __launch_bounds__(WAVES * 32)
void egnn_node_kernel(const float* __restrict__ h,
                      const float* __restrict__ agg,
                      const float* __restrict__ nb1,
                      const float* __restrict__ nb2,
                      const _Float16* __restrict__ packN1,
                      const _Float16* __restrict__ packN2,
                      float* __restrict__ outH,
                      int N, int nTiles) {
    __shared__ __align__(16) _Float16 sIn[WAVES][16][2 * F];
    __shared__ __align__(16) _Float16 sHid[WAVES][16][H];

    const int lane = threadIdx.x & 31;
    const int w    = threadIdx.x >> 5;
    const int tile = blockIdx.x * WAVES + w;
    const bool valid = tile < nTiles;
    const int n0 = tile * 16;
    const int hiM = (lane >> 4) * 8;
    const int nl  = lane & 15;

    // ---- gather concat(h, agg) rows (sequential node ids) ----
    if (valid) {
#pragma unroll 4
        for (int m = 0; m < 16; ++m) {
            int node = n0 + m;
            v4h za, zb;
            if (node < N) {
                float4 a = ((const float4*)(h   + (size_t)node * F))[lane];
                float4 g = ((const float4*)(agg + (size_t)node * H))[lane];
                za = (v4h){(_Float16)a.x, (_Float16)a.y, (_Float16)a.z, (_Float16)a.w};
                zb = (v4h){(_Float16)g.x, (_Float16)g.y, (_Float16)g.z, (_Float16)g.w};
            } else {
                za = (v4h)(_Float16)0.0f;
                zb = za;
            }
            *(v4h*)&sIn[w][m][lane * 4]     = za;
            *(v4h*)&sIn[w][m][F + lane * 4] = zb;
        }
    }
    __syncthreads();

    // ---- node GEMM1 (K=256) + bias + SiLU ----
    if (valid) {
        v8f acc[8];
#pragma unroll
        for (int nt = 0; nt < 8; ++nt) acc[nt] = (v8f)0.0f;
        wave_gemm<8>(&sIn[w][0][0], 2 * F, packN1, acc, lane);
#pragma unroll
        for (int nt = 0; nt < 8; ++nt) {
            int n = nt * 16 + nl;
            float b1 = nb1[n];
#pragma unroll
            for (int j = 0; j < 8; ++j) {
                int m = j + hiM;
                sHid[w][m][n] = (_Float16)silu_f(acc[nt][j] + b1);
            }
        }
    }
    __syncthreads();

    // ---- node GEMM2 (K=128) + bias + residual -> h_new ----
    if (valid) {
        v8f acc[8];
#pragma unroll
        for (int nt = 0; nt < 8; ++nt) acc[nt] = (v8f)0.0f;
        wave_gemm<4>(&sHid[w][0][0], H, packN2, acc, lane);
#pragma unroll
        for (int nt = 0; nt < 8; ++nt) {
            int n = nt * 16 + nl;
            float b2 = nb2[n];
#pragma unroll
            for (int j = 0; j < 8; ++j) {
                int m = j + hiM;
                int node = n0 + m;
                if (node < N) {
                    size_t idx = (size_t)node * F + n;
                    outH[idx] = h[idx] + acc[nt][j] + b2;
                }
            }
        }
    }
}

// ---------------------------------------------------------------------------
extern "C" void kernel_launch(void* const* d_in, const int* in_sizes, int n_in,
                              void* d_out, int out_size, void* d_ws, size_t ws_size,
                              hipStream_t stream) {
    const float* h         = (const float*)d_in[0];
    const float* pos       = (const float*)d_in[1];
    const int*   senders   = (const int*)d_in[2];
    const int*   receivers = (const int*)d_in[3];
    const float* eW1 = (const float*)d_in[4];
    const float* eb1 = (const float*)d_in[5];
    const float* eW2 = (const float*)d_in[6];
    const float* eb2 = (const float*)d_in[7];
    const float* nW1 = (const float*)d_in[8];
    const float* nb1 = (const float*)d_in[9];
    const float* nW2 = (const float*)d_in[10];
    const float* nb2 = (const float*)d_in[11];
    const float* pW1 = (const float*)d_in[12];
    const float* pb1 = (const float*)d_in[13];
    const float* pW2 = (const float*)d_in[14];

    const int N = in_sizes[0] / F;
    const int E = in_sizes[2];

    float* agg = (float*)d_ws;
    _Float16* pk = (_Float16*)((char*)d_ws + (size_t)N * F * sizeof(float));
    _Float16* packE1 = pk;                 // 256*128 halfs
    _Float16* packE2 = pk + 32768;         // 128*128
    _Float16* packN1 = pk + 49152;         // 256*128
    _Float16* packN2 = pk + 81920;         // 128*128
    _Float16* packP1 = pk + 98304;         // 128*128

    float* outH   = (float*)d_out;
    float* outPos = outH + (size_t)N * F;

    // init: zero agg, seed pos_out = pos
    {
        int tot = N * F;
        egnn_init_kernel<<<(tot + 255) / 256, 256, 0, stream>>>(agg, outPos, pos, N);
    }
    // pack weights to WMMA-B f16 fragments
    egnn_pack_w16<<<(256 * 128 + 255) / 256, 256, 0, stream>>>(eW1, packE1, 256);
    egnn_pack_w16<<<(128 * 128 + 255) / 256, 256, 0, stream>>>(eW2, packE2, 128);
    egnn_pack_w16<<<(256 * 128 + 255) / 256, 256, 0, stream>>>(nW1, packN1, 256);
    egnn_pack_w16<<<(128 * 128 + 255) / 256, 256, 0, stream>>>(nW2, packN2, 128);
    egnn_pack_w16<<<(128 * 128 + 255) / 256, 256, 0, stream>>>(pW1, packP1, 128);

    // fused edge pipeline
    {
        int eTiles = (E + 15) / 16;
        int blocks = (eTiles + WAVES - 1) / WAVES;
        egnn_edge_kernel<<<blocks, WAVES * 32, 0, stream>>>(
            h, pos, senders, receivers, eW1, eb1, eb2, pb1, pW2,
            packE1, packE2, packP1, agg, outPos, E, eTiles);
    }
    // node pipeline
    {
        int nTiles = (N + 15) / 16;
        int blocks = (nTiles + WAVES - 1) / WAVES;
        egnn_node_kernel<<<blocks, WAVES * 32, 0, stream>>>(
            h, agg, nb1, nb2, packN1, packN2, outH, N, nTiles);
    }
}
